// Transformer_52338471469301
// MI455X (gfx1250) — matrix operand order repaired and checked by reading, not compiled, not run on hardware
//
#include <hip/hip_runtime.h>

#define BB 64
#define FF 256
#define TT 1024
#define DD 512

typedef __attribute__((ext_vector_type(16))) __bf16 v16bf;
typedef __attribute__((ext_vector_type(8)))  float  v8f;

union Frag {
  v16bf v;
  uint4 q[2];
};

__device__ __forceinline__ unsigned short f2bf(float f) {
  unsigned int u = __builtin_bit_cast(unsigned int, f);
  u += 0x7FFFu + ((u >> 16) & 1u);
  return (unsigned short)(u >> 16);
}

__device__ __forceinline__ v8f wmma_bf16(v16bf a, v16bf b, v8f c) {
  return __builtin_amdgcn_wmma_f32_16x16x32_bf16(false, a, false, b, (short)0, c,
                                                 false, false);
}

// A-fragment (16x32 bf16, MxK) from an LDS tile, row-major with stride ld.
// ISA layout: lane<16 -> M=lane, K in {0..7,16..23}; lane>=16 -> K+8.
__device__ __forceinline__ v16bf load_A_lds(const unsigned short* base, int ld, int k0) {
  int lane = threadIdx.x & 31;
  int row = lane & 15, kh = lane >> 4;
  const unsigned short* p = base + row * ld + k0 + kh * 8;
  Frag f;
  f.q[0] = *(const uint4*)(p);
  f.q[1] = *(const uint4*)(p + 16);
  return f.v;
}

// B-fragment (32x16 bf16, KxN) where B(k,n) = M[n0+n, k] for row-major M[N,K]
// (i.e. computing X @ M^T). Two 16B reads per lane: K in {0..7,16..23}, lane>=16 -> K+8.
__device__ __forceinline__ v16bf load_B_glb(const unsigned short* base, int ld, int n0, int k0) {
  int lane = threadIdx.x & 31;
  int col = lane & 15, kh = lane >> 4;
  const unsigned short* p = base + (size_t)(n0 + col) * ld + k0 + kh * 8;
  Frag f;
  f.q[0] = *(const uint4*)(p);
  f.q[1] = *(const uint4*)(p + 16);
  return f.v;
}

__global__ void cvt_bf16_kernel(const float* __restrict__ in,
                                unsigned short* __restrict__ out, int n) {
  int i = blockIdx.x * blockDim.x + threadIdx.x;
  if (i < n) out[i] = f2bf(in[i]);
}

__global__ void zero_kernel(float* p) { *p = 0.0f; }

// gate[b,l] = sigmoid( X[b,l,:].Wg + Pos[l,:].Wgp + bg + bgp + extra )
__global__ void gate_kernel(const float* __restrict__ X, const float* __restrict__ Pos,
                            const float* __restrict__ Wg, const float* __restrict__ Wgp,
                            const float* __restrict__ bg, const float* __restrict__ bgp,
                            float extra, float* __restrict__ out, int L, int nrows) {
  int wid = (int)((blockIdx.x * blockDim.x + threadIdx.x) >> 5);
  if (wid >= nrows) return;
  int lane = threadIdx.x & 31;
  int b = wid / L, l = wid % L;
  const float* xr = X + ((size_t)b * L + l) * DD;
  const float* pr = Pos + (size_t)l * DD;
  float s = 0.f;
  for (int i = lane; i < DD; i += 32) s += xr[i] * Wg[i] + pr[i] * Wgp[i];
  for (int o = 16; o >= 1; o >>= 1) s += __shfl_xor(s, o, 32);
  if (lane == 0) out[wid] = 1.0f / (1.0f + __expf(-(s + bg[0] + bgp[0] + extra)));
}

#define ALD 520  // bf16 LDS tile row stride (16B-aligned, bank-spread)
#define OLD 516  // f32 LDS tile row stride

// Y[b, row, :] = norm( X[b,row,:] @ W1^T + Pos[row,:] @ W2^T + b1 + b2 )  (do_norm=1)
// or transposed un-normalized bf16 out Y[b, d, row] (V path, do_norm=0).
__global__ void proj_kernel(const float* __restrict__ X, const float* __restrict__ Pos,
                            const unsigned short* __restrict__ W1,
                            const unsigned short* __restrict__ W2,
                            const float* __restrict__ b1, const float* __restrict__ b2,
                            unsigned short* __restrict__ Y, int L, int do_norm) {
  extern __shared__ char smem[];
  float* Ot = (float*)smem;                              // 16 x 516 f32
  unsigned short* At = (unsigned short*)(smem + 33024);  // 16 x 520 bf16
  unsigned short* Pt = (unsigned short*)(smem + 49664);  // 16 x 520 bf16
  int tiles = L >> 4;
  int b = blockIdx.x / tiles;
  int row0 = (blockIdx.x % tiles) << 4;
  int tid = threadIdx.x;

  // Stage A-tile (f32 -> bf16) into LDS.
  const float* src = X + ((size_t)b * L + row0) * DD;
  for (int it = tid; it < 16 * 128; it += 256) {
    int r = it >> 7, c4 = it & 127;
    float4 v = ((const float4*)(src + (size_t)r * DD))[c4];
    uint2 pk;
    pk.x = (unsigned int)f2bf(v.x) | ((unsigned int)f2bf(v.y) << 16);
    pk.y = (unsigned int)f2bf(v.z) | ((unsigned int)f2bf(v.w) << 16);
    *(uint2*)(At + r * ALD + (c4 << 2)) = pk;
  }
  if (Pos) {
    const float* ps = Pos + (size_t)row0 * DD;
    for (int it = tid; it < 16 * 128; it += 256) {
      int r = it >> 7, c4 = it & 127;
      float4 v = ((const float4*)(ps + (size_t)r * DD))[c4];
      uint2 pk;
      pk.x = (unsigned int)f2bf(v.x) | ((unsigned int)f2bf(v.y) << 16);
      pk.y = (unsigned int)f2bf(v.z) | ((unsigned int)f2bf(v.w) << 16);
      *(uint2*)(Pt + r * ALD + (c4 << 2)) = pk;
    }
  }
  __syncthreads();

  int w = tid >> 5, lane = tid & 31, col = lane & 15, rh = lane >> 4;
  int n0 = w << 6;  // wave owns N columns [n0, n0+64): 4 tiles sharing one A-frag
  v8f acc[4] = {v8f{}, v8f{}, v8f{}, v8f{}};
#pragma unroll
  for (int kk = 0; kk < DD; kk += 32) {
    v16bf a = load_A_lds(At, ALD, kk);
#pragma unroll
    for (int j = 0; j < 4; ++j)
      acc[j] = wmma_bf16(a, load_B_glb(W1, DD, n0 + (j << 4), kk), acc[j]);
  }
  if (Pos) {
#pragma unroll
    for (int kk = 0; kk < DD; kk += 32) {
      v16bf a = load_A_lds(Pt, ALD, kk);
#pragma unroll
      for (int j = 0; j < 4; ++j)
        acc[j] = wmma_bf16(a, load_B_glb(W2, DD, n0 + (j << 4), kk), acc[j]);
    }
  }
#pragma unroll
  for (int j = 0; j < 4; ++j) {
    int n = n0 + (j << 4) + col;
    float bias = b1[n] + (b2 ? b2[n] : 0.0f);
#pragma unroll
    for (int i = 0; i < 8; ++i)
      Ot[((rh << 3) + i) * OLD + n] = acc[j][i] + bias;
  }
  __syncthreads();

  int row = tid >> 4, l16 = tid & 15;
  if (do_norm) {
    float s = 0.f;
    for (int i = 0; i < DD / 16; ++i) {
      float x = Ot[row * OLD + l16 + i * 16];
      s += x * x;
    }
    for (int o = 8; o >= 1; o >>= 1) s += __shfl_xor(s, o, 16);
    float inv = 1.0f / fmaxf(sqrtf(s), 1e-12f);
    unsigned short* dst = Y + ((size_t)b * L + row0 + row) * DD;
    for (int i = 0; i < DD / 16; ++i) {
      int c = l16 + i * 16;
      dst[c] = f2bf(Ot[row * OLD + c] * inv);
    }
  } else {
    // transposed bf16 out: Y[b, d, row]
    for (int i = 0; i < DD / 16; ++i) {
      int c = l16 + i * 16;
      Y[((size_t)b * DD + c) * L + row0 + row] = f2bf(Ot[row * OLD + c]);
    }
  }
}

#define SLD 1028  // f32 score-row LDS stride

// One block = one (b, 16-row F tile): full score row, qloss max, softmax, gates, probs.
__global__ void attn_kernel(const unsigned short* __restrict__ Qbf,
                            const unsigned short* __restrict__ Kbf,
                            const float* __restrict__ gt, const float* __restrict__ gf,
                            float* __restrict__ probs_out,
                            unsigned short* __restrict__ probs_bf,
                            float* __restrict__ qloss) {
  extern __shared__ char smem[];
  unsigned short* Qt = (unsigned short*)smem;   // 16 x 520 bf16
  float* S = (float*)(smem + 16640);            // 16 x 1028 f32
  int b = blockIdx.x / (FF >> 4);
  int row0 = (blockIdx.x % (FF >> 4)) << 4;
  int tid = threadIdx.x;

  const uint4* qsrc = (const uint4*)(Qbf + ((size_t)b * FF + row0) * DD);
  for (int it = tid; it < 16 * 64; it += 256) {
    int r = it >> 6, c = it & 63;
    ((uint4*)(Qt + r * ALD))[c] = qsrc[r * 64 + c];
  }
  __syncthreads();

  int w = tid >> 5, lane = tid & 31, col = lane & 15, rh = lane >> 4;
  const unsigned short* Kb = Kbf + (size_t)b * TT * DD;
  // Wave owns T columns [w*128, w*128+128): 2 passes of 4 tiles sharing A-frags.
#pragma unroll
  for (int pass = 0; pass < 2; ++pass) {
    int t0 = (w << 7) + (pass << 6);
    v8f acc[4] = {v8f{}, v8f{}, v8f{}, v8f{}};
#pragma unroll
    for (int kk = 0; kk < DD; kk += 32) {
      v16bf a = load_A_lds(Qt, ALD, kk);
#pragma unroll
      for (int j = 0; j < 4; ++j)
        acc[j] = wmma_bf16(a, load_B_glb(Kb, DD, t0 + (j << 4), kk), acc[j]);
    }
#pragma unroll
    for (int j = 0; j < 4; ++j)
#pragma unroll
      for (int i = 0; i < 8; ++i)
        S[((rh << 3) + i) * SLD + t0 + (j << 4) + col] = acc[j][i];
  }
  __syncthreads();

  int row = tid >> 4, l16 = tid & 15;
  float m = -3.4e38f;
  for (int i = 0; i < TT / 16; ++i) m = fmaxf(m, S[row * SLD + l16 + i * 16]);
  for (int o = 8; o >= 1; o >>= 1) m = fmaxf(m, __shfl_xor(m, o, 16));
  if (l16 == 0)
    atomicAdd(qloss, (1.0f - m) * (1.0f / ((float)BB * (float)FF)));

  const float SC = 0.44194173824159216f;  // 1/(sqrt(512)*0.1)
  float s = 0.f;
  for (int i = 0; i < TT / 16; ++i) {
    int t = l16 + i * 16;
    float e = __expf(SC * (S[row * SLD + t] - m));
    S[row * SLD + t] = e;
    s += e;
  }
  for (int o = 8; o >= 1; o >>= 1) s += __shfl_xor(s, o, 16);

  float inv = gf[(size_t)b * FF + row0 + row] / s;
  float* po = probs_out + ((size_t)b * FF + row0 + row) * TT;
  unsigned short* pb = probs_bf + ((size_t)b * FF + row0 + row) * TT;
  const float* gtb = gt + (size_t)b * TT;
  for (int i = 0; i < TT / 16; ++i) {
    int t = l16 + i * 16;
    float p = S[row * SLD + t] * inv * gtb[t];
    po[t] = p;
    pb[t] = f2bf(p);
  }
}

#define PLD 1032  // probs bf16 LDS stride

// control = probs @ V (via V^T), then out = control @ Wo^T + bo
__global__ void ctrl_kernel(const unsigned short* __restrict__ Pbf,
                            const unsigned short* __restrict__ VT,
                            const unsigned short* __restrict__ Wo,
                            const float* __restrict__ bo, float* __restrict__ out) {
  extern __shared__ char smem[];
  unsigned short* Pt = (unsigned short*)smem;            // 16 x 1032 bf16
  unsigned short* Cb = (unsigned short*)(smem + 33024);  // 16 x 520 bf16
  int b = blockIdx.x >> 4;           // FF/16 == 16
  int row0 = (blockIdx.x & 15) << 4;
  int tid = threadIdx.x;

  const uint4* psrc = (const uint4*)(Pbf + ((size_t)b * FF + row0) * TT);
  for (int it = tid; it < 16 * 128; it += 256) {
    int r = it >> 7, c = it & 127;
    ((uint4*)(Pt + r * PLD))[c] = psrc[r * 128 + c];
  }
  __syncthreads();

  int w = tid >> 5, lane = tid & 31, col = lane & 15, rh = lane >> 4;
  int n0 = w << 6;  // wave owns D columns [n0, n0+64)
  const unsigned short* Vb = VT + (size_t)b * DD * TT;
  {
    v8f acc[4] = {v8f{}, v8f{}, v8f{}, v8f{}};
#pragma unroll
    for (int kk = 0; kk < TT; kk += 32) {
      v16bf a = load_A_lds(Pt, PLD, kk);
#pragma unroll
      for (int j = 0; j < 4; ++j)
        acc[j] = wmma_bf16(a, load_B_glb(Vb, TT, n0 + (j << 4), kk), acc[j]);
    }
#pragma unroll
    for (int j = 0; j < 4; ++j)
#pragma unroll
      for (int i = 0; i < 8; ++i)
        Cb[((rh << 3) + i) * ALD + n0 + (j << 4) + col] = f2bf(acc[j][i]);
  }
  __syncthreads();
  {
    v8f acc[4] = {v8f{}, v8f{}, v8f{}, v8f{}};
#pragma unroll
    for (int kk = 0; kk < DD; kk += 32) {
      v16bf a = load_A_lds(Cb, ALD, kk);
#pragma unroll
      for (int j = 0; j < 4; ++j)
        acc[j] = wmma_bf16(a, load_B_glb(Wo, DD, n0 + (j << 4), kk), acc[j]);
    }
#pragma unroll
    for (int j = 0; j < 4; ++j) {
      int n = n0 + (j << 4) + col;
      float bias = bo[n];
#pragma unroll
      for (int i = 0; i < 8; ++i) {
        int r = (rh << 3) + i;
        out[((size_t)(b * FF) + row0 + r) * DD + n] = acc[j][i] + bias;
      }
    }
  }
}

extern "C" void kernel_launch(void* const* d_in, const int* in_sizes, int n_in,
                              void* d_out, int out_size, void* d_ws, size_t ws_size,
                              hipStream_t stream) {
  (void)in_sizes; (void)n_in; (void)out_size; (void)ws_size;
  const float* from_tensor = (const float*)d_in[0];
  const float* to_tensor   = (const float*)d_in[1];
  const float* from_pos    = (const float*)d_in[2];
  const float* to_pos      = (const float*)d_in[3];
  const float* Wq  = (const float*)d_in[4];  const float* bq  = (const float*)d_in[5];
  const float* Wk  = (const float*)d_in[6];  const float* bk  = (const float*)d_in[7];
  const float* Wv  = (const float*)d_in[8];  const float* bv  = (const float*)d_in[9];
  const float* Wfp = (const float*)d_in[10]; const float* bfp = (const float*)d_in[11];
  const float* Wtp = (const float*)d_in[12]; const float* btp = (const float*)d_in[13];
  const float* Wgt = (const float*)d_in[14]; const float* bgt = (const float*)d_in[15];
  const float* Wgtp= (const float*)d_in[16]; const float* bgtp= (const float*)d_in[17];
  const float* Wgf = (const float*)d_in[18]; const float* bgf = (const float*)d_in[19];
  const float* Wgfp= (const float*)d_in[20]; const float* bgfp= (const float*)d_in[21];
  const float* Wo  = (const float*)d_in[22]; const float* bo  = (const float*)d_in[23];

  char* ws = (char*)d_ws;
  unsigned short* QBF  = (unsigned short*)(ws + 0);          // B*F*D bf16
  unsigned short* KBF  = (unsigned short*)(ws + 16777216);   // B*T*D bf16
  unsigned short* VTBF = (unsigned short*)(ws + 83886080);   // B*D*T bf16 (transposed)
  unsigned short* PRBF = (unsigned short*)(ws + 150994944);  // B*F*T bf16
  unsigned short* WQB  = (unsigned short*)(ws + 184549376);
  unsigned short* WKB  = (unsigned short*)(ws + 185073664);
  unsigned short* WVB  = (unsigned short*)(ws + 185597952);
  unsigned short* WOB  = (unsigned short*)(ws + 186122240);
  unsigned short* WFPB = (unsigned short*)(ws + 186646528);
  unsigned short* WTPB = (unsigned short*)(ws + 187170816);
  float* GT = (float*)(ws + 187695104);  // B*T f32
  float* GF = (float*)(ws + 187957248);  // B*F f32

  float* out_ptr   = (float*)d_out;
  float* probs_ptr = out_ptr + (size_t)BB * FF * DD;
  float* qloss_ptr = probs_ptr + (size_t)BB * FF * TT;

  // Weight f32 -> bf16
  cvt_bf16_kernel<<<1024, 256, 0, stream>>>(Wq,  WQB,  DD * DD);
  cvt_bf16_kernel<<<1024, 256, 0, stream>>>(Wk,  WKB,  DD * DD);
  cvt_bf16_kernel<<<1024, 256, 0, stream>>>(Wv,  WVB,  DD * DD);
  cvt_bf16_kernel<<<1024, 256, 0, stream>>>(Wo,  WOB,  DD * DD);
  cvt_bf16_kernel<<<1024, 256, 0, stream>>>(Wfp, WFPB, DD * DD);
  cvt_bf16_kernel<<<1024, 256, 0, stream>>>(Wtp, WTPB, DD * DD);

  // Q/K (fused pos-proj + bias + l2norm), V (transposed bf16)
  proj_kernel<<<BB * (FF / 16), 256, 66304, stream>>>(from_tensor, from_pos, WQB, WFPB,
                                                      bq, bfp, QBF, FF, 1);
  proj_kernel<<<BB * (TT / 16), 256, 66304, stream>>>(to_tensor, to_pos, WKB, WTPB,
                                                      bk, btp, KBF, TT, 1);
  proj_kernel<<<BB * (TT / 16), 256, 66304, stream>>>(to_tensor, nullptr, WVB, nullptr,
                                                      bv, nullptr, VTBF, TT, 0);
  // Gates
  gate_kernel<<<(BB * TT) / 8, 256, 0, stream>>>(to_tensor, to_pos, Wgt, Wgtp, bgt, bgtp,
                                                 0.0f, GT, TT, BB * TT);
  gate_kernel<<<(BB * FF) / 8, 256, 0, stream>>>(from_tensor, from_pos, Wgf, Wgfp, bgf, bgfp,
                                                 1.0f, GF, FF, BB * FF);
  zero_kernel<<<1, 1, 0, stream>>>(qloss_ptr);
  // Scores + qloss + softmax + gates -> probs
  attn_kernel<<<BB * (FF / 16), 256, 82432, stream>>>(QBF, KBF, GT, GF, probs_ptr, PRBF,
                                                      qloss_ptr);
  // probs @ V, then @ Wo^T + bo -> out
  ctrl_kernel<<<BB * (FF / 16), 256, 49664, stream>>>(PRBF, VTBF, WOB, bo, out_ptr);
}
